// YoloLayer_9088150798344
// MI455X (gfx1250) — compile-verified, matrix operand order checked by reading, not run on hardware
//
#include <hip/hip_runtime.h>

// ============================================================================
// YOLO post-process for MI455X (gfx1250, wave32).
//  K1 decode     : fused sigmoid/exp anchor decode; writes boxes [B,N,4] and
//                  class scores TRANSPOSED [B,C,N] so later scans are unit-stride.
//  K2 topk       : per (b,c) block; score row (100KB) streamed through LDS with
//                  the Tensor Data Mover (double buffered on TENSORcnt);
//                  histogram threshold -> compact -> bitonic sort top-256.
//  K3 nms        : per (b,c) block; 256x256 pairwise IoU built as 16x16 tiles;
//                  area_i+area_j rank-2 outer sum on the matrix pipe via
//                  v_wmma_f32_16x16x32_f16, intersection on VALU; adjacency
//                  bitmasks + serial greedy suppression.
//  K4 final      : per image; top-100 over 80*256 kept scores, clip boxes,
//                  emit nb|nc|ns|num_detections.
// Everything is bandwidth-bound (~265MB total traffic -> ~11us at 23.3TB/s).
// ============================================================================

#define NB    8
#define NCLS  80
#define NPTS  25200          // 80*80*3 + 40*40*3 + 20*20*3
#define KSEL  256            // pre-NMS candidates per class
#define KEEP  100            // final boxes per image
#define CH    6400           // LDS staging chunk (floats) for topk
#define NMS_T 0.6f

typedef _Float16     v16h __attribute__((ext_vector_type(16)));
typedef float        v8f  __attribute__((ext_vector_type(8)));
typedef unsigned int v4u  __attribute__((ext_vector_type(4)));
typedef int          v8i  __attribute__((ext_vector_type(8)));
typedef int          v4i  __attribute__((ext_vector_type(4)));

__constant__ float g_anchors[18] = {
    12, 16,  19, 36,  40, 28,     // level 3
    36, 75,  76, 55,  72, 146,    // level 4
    142,110, 192,243, 459,401 };  // level 5

__device__ __forceinline__ float sigm(float x) { return 1.0f / (1.0f + __expf(-x)); }

#if defined(__has_builtin)
#if __has_builtin(__builtin_amdgcn_tensor_load_to_lds)
#define HAVE_TDM 1
#endif
#endif

#ifdef HAVE_TDM
// Low 32 bits of a flat shared-aperture address are the LDS byte offset
// (ISA 10.2).
__device__ __forceinline__ unsigned lds_addr_of(const void* p) {
    return (unsigned)(unsigned long long)p;
}

// 1-D TDM tile load: `elems` f32 from global -> LDS. D# per CDNA5 ISA.
__device__ __forceinline__ void tdm_load_row(const float* gsrc, unsigned lds_byte,
                                             unsigned elems) {
    unsigned long long ga = (unsigned long long)gsrc;
    v4u g0 = { 1u,                                              // count=1 (valid user D#)
               lds_byte,                                        // lds_addr [63:32]
               (unsigned)(ga & 0xffffffffull),                  // global_addr lo
               (unsigned)((ga >> 32) & 0x1ffffffull)            // global_addr [120:96]
                   | 0x80000000u };                             // type=2 ("image")
    v8i g1 = { (int)(2u << 16),                                 // data_size=4B, wgmask=0
               (int)((elems & 0xffffu) << 16),                  // tensor_dim0[15:0]
               (int)(((elems >> 16) & 0xffffu) | (1u << 16)),   // dim0[31:16] | tensor_dim1=1
               (int)((elems & 0xffffu) << 16),                  // tile_dim0 = elems
               0,                                               // tile_dim1/2 unused
               (int)elems,                                      // tensor_dim0_stride
               0, 0 };
    v4i z4 = {};
#if __clang_major__ >= 23
    v8i z8 = {};
    __builtin_amdgcn_tensor_load_to_lds(g0, g1, z4, z4, z8, 0);
#else
    __builtin_amdgcn_tensor_load_to_lds(g0, g1, z4, z4, 0);
#endif
}
#endif

// Stream one score row (NPTS f32) through LDS in CH-sized chunks and hand each
// staged chunk to `consume`. TDM double-buffered: chunk c+1 in flight while
// chunk c is consumed; wave 0 issues, TENSORcnt gates, barriers publish.
template <typename F>
__device__ void scan_row_chunks(const float* __restrict__ row,
                                float* buf0, float* buf1, F&& consume) {
    const int nch = (NPTS + CH - 1) / CH;
#ifdef HAVE_TDM
    const bool issuer = (threadIdx.x < 32);   // wave 0 only
    if (issuer) tdm_load_row(row, lds_addr_of(buf0), (NPTS < CH) ? NPTS : CH);
    for (int c = 0; c < nch; ++c) {
        float* cur = (c & 1) ? buf1 : buf0;
        if (issuer) {
            if (c + 1 < nch) {
                int ne = NPTS - (c + 1) * CH; if (ne > CH) ne = CH;
                tdm_load_row(row + (size_t)(c + 1) * CH,
                             lds_addr_of(((c + 1) & 1) ? buf1 : buf0), (unsigned)ne);
                __builtin_amdgcn_s_wait_tensorcnt(1);   // chunk c complete, c+1 in flight
            } else {
                __builtin_amdgcn_s_wait_tensorcnt(0);
            }
        }
        __syncthreads();
        int elems = NPTS - c * CH; if (elems > CH) elems = CH;
        consume(cur, c * CH, elems);
        __syncthreads();                                // before buffer reuse
    }
#else
    (void)buf1;
    for (int c = 0; c < nch; ++c) {
        int elems = NPTS - c * CH; if (elems > CH) elems = CH;
        for (int e = threadIdx.x; e < elems; e += blockDim.x) buf0[e] = row[c * CH + e];
        __syncthreads();
        consume(buf0, c * CH, elems);
        __syncthreads();
    }
#endif
}

// ------------------------------- K1: decode --------------------------------
__global__ __launch_bounds__(256) void decode_kernel(
        const float* __restrict__ x3, const float* __restrict__ x4,
        const float* __restrict__ x5, float* __restrict__ boxes,
        float* __restrict__ clsT) {
    int tid = blockIdx.x * blockDim.x + threadIdx.x;
    if (tid >= NB * NPTS) return;
    const int b = tid / NPTS;
    const int n = tid % NPTS;

    const float* lp; int m, W, lvl; size_t bstride;
    if (n < 19200)      { lvl = 0; m = n;         W = 80; lp = x3; bstride = 1632000u; }
    else if (n < 24000) { lvl = 1; m = n - 19200; W = 40; lp = x4; bstride = 408000u;  }
    else                { lvl = 2; m = n - 24000; W = 20; lp = x5; bstride = 102000u;  }
    const int a  = m % 3;
    const int hw = m / 3;
    const int wc = hw % W, hc = hw / W;
    const float* src = lp + (size_t)b * bstride + (size_t)hw * 255 + a * 85;

    const float aw = g_anchors[lvl * 6 + 2 * a]     * (1.0f / 640.0f);
    const float ah = g_anchors[lvl * 6 + 2 * a + 1] * (1.0f / 640.0f);
    const float invW = 1.0f / (float)W;            // H == W on every level

    const float xc = (sigm(src[0]) + (float)wc) * invW;
    const float yc = (sigm(src[1]) + (float)hc) * invW;
    const float bw = __expf(src[2]) * aw;
    const float bh = __expf(src[3]) * ah;
    const float obj = sigm(src[4]);

    const size_t bo = ((size_t)b * NPTS + n) * 4;
    boxes[bo + 0] = yc - 0.5f * bh;
    boxes[bo + 1] = xc - 0.5f * bw;
    boxes[bo + 2] = yc + 0.5f * bh;
    boxes[bo + 3] = xc + 0.5f * bw;

    // transposed scores: consecutive n within a wave -> coalesced per class
    for (int c = 0; c < NCLS; ++c)
        clsT[((size_t)(b * NCLS + c)) * NPTS + n] = sigm(src[5 + c]) * obj;
}

// ------------------------------- K2: top-256 -------------------------------
__global__ __launch_bounds__(256) void topk_kernel(
        const float* __restrict__ clsT, const float* __restrict__ boxes,
        float* __restrict__ cand_val, float* __restrict__ cand_box) {
    __shared__ float    buf0[CH];
    __shared__ float    buf1[CH];
    __shared__ unsigned hist[1024];
    __shared__ float    selv[KSEL];
    __shared__ int      seli[KSEL];
    __shared__ unsigned cnt_hi;
    __shared__ int      s_T, s_C1, s_tie;

    const int bc  = blockIdx.x;            // b*NCLS + c
    const int tid = threadIdx.x;
    const float* row = clsT + (size_t)bc * NPTS;

    for (int i = tid; i < 1024; i += 256) hist[i] = 0u;
    if (tid == 0) { cnt_hi = 0u; s_tie = 0; }
    __syncthreads();

    // pass 1: value histogram (scores live in [0,1))
    scan_row_chunks(row, buf0, buf1, [&](const float* p, int base, int elems) {
        (void)base;
        for (int e = tid; e < elems; e += 256) {
            int bin = (int)(p[e] * 1024.0f);
            bin = bin < 0 ? 0 : (bin > 1023 ? 1023 : bin);
            atomicAdd(&hist[bin], 1u);
        }
    });

    if (tid == 0) {                        // threshold bin for K=256
        unsigned acc = 0; int T = 0, C1 = 0;
        for (int t = 1023; t >= 0; --t) {
            unsigned h = hist[t];
            if (acc + h >= KSEL) { T = t; C1 = (int)acc; break; }
            acc += h;
        }
        s_T = T; s_C1 = C1;
    }
    __syncthreads();
    const int T = s_T, C1 = s_C1, need = KSEL - C1;
    for (int i = tid; i < KSEL; i += 256) { selv[i] = -1.0f; seli[i] = 0; }
    __syncthreads();

    // pass 2: compact.  bin>T set is exact (size C1) -> parallel atomics into
    // slots [0,C1); ties at bin==T filled serially in index order (deterministic).
    scan_row_chunks(row, buf0, buf1, [&](const float* p, int base, int elems) {
        for (int e = tid; e < elems; e += 256) {
            float v = p[e];
            int bin = (int)(v * 1024.0f);
            bin = bin < 0 ? 0 : (bin > 1023 ? 1023 : bin);
            if (bin > T) {
                unsigned s = atomicAdd(&cnt_hi, 1u);
                selv[s] = v; seli[s] = base + e;
            }
        }
        if (tid == 0) {
            int t = s_tie;
            for (int e = 0; e < elems && t < need; ++e) {
                float v = p[e];
                int bin = (int)(v * 1024.0f);
                bin = bin < 0 ? 0 : (bin > 1023 ? 1023 : bin);
                if (bin == T) { selv[C1 + t] = v; seli[C1 + t] = base + e; ++t; }
            }
            s_tie = t;
        }
    });

    // bitonic sort: descending value, ascending index on ties
    for (int k = 2; k <= KSEL; k <<= 1)
        for (int j = k >> 1; j > 0; j >>= 1) {
            __syncthreads();
            int ixj = tid ^ j;
            if (ixj > tid) {
                bool desc = ((tid & k) == 0);
                float va = selv[tid], vb = selv[ixj];
                int   ia = seli[tid], ib = seli[ixj];
                bool swp = desc ? ((va < vb) || (va == vb && ia > ib))
                                : ((va > vb) || (va == vb && ia < ib));
                if (swp) { selv[tid] = vb; selv[ixj] = va;
                           seli[tid] = ib; seli[ixj] = ia; }
            }
        }
    __syncthreads();

    const int b = bc / NCLS;
    const float v = selv[tid];
    const int   n = seli[tid];
    const size_t co = (size_t)bc * KSEL + tid;
    cand_val[co] = v > 0.0f ? v : 0.0f;
    const size_t sb = ((size_t)b * NPTS + n) * 4;
    const size_t db = co * 4;
    cand_box[db + 0] = boxes[sb + 0];
    cand_box[db + 1] = boxes[sb + 1];
    cand_box[db + 2] = boxes[sb + 2];
    cand_box[db + 3] = boxes[sb + 3];
}

// ------------------------------- K3: NMS -----------------------------------
__global__ __launch_bounds__(256) void nms_kernel(
        float* __restrict__ cand_val, const float* __restrict__ cand_box) {
    __shared__ float    y0s[KSEL], x0s[KSEL], y1s[KSEL], x1s[KSEL], scs[KSEL];
    __shared__ _Float16 areaH[KSEL];
    __shared__ unsigned adj[KSEL][8];      // adjacency bitmask: iou(i,j) > 0.6
    __shared__ unsigned keepm[8];

    const int bc  = blockIdx.x;
    const int tid = threadIdx.x;
    const size_t base = (size_t)bc * KSEL;

    const float b0 = cand_box[(base + tid) * 4 + 0];
    const float b1 = cand_box[(base + tid) * 4 + 1];
    const float b2 = cand_box[(base + tid) * 4 + 2];
    const float b3 = cand_box[(base + tid) * 4 + 3];
    y0s[tid] = b0; x0s[tid] = b1; y1s[tid] = b2; x1s[tid] = b3;
    areaH[tid] = (_Float16)((b2 - b0) * (b3 - b1));
    scs[tid] = cand_val[base + tid];
    for (int w = 0; w < 8; ++w) adj[tid][w] = 0u;
    __syncthreads();

    // 16x16 tiles of the 256x256 pair matrix; 8 waves x 32 tiles.
    // Matrix pipe computes the rank-2 outer sum area_i + area_j:
    //   A rows = [area_i, 1, 0...0], B cols = [1, area_j, 0...0]^T
    const int wave = tid >> 5, lane = tid & 31;
    for (int t = wave * 32; t < wave * 32 + 32; ++t) {
        const int i0 = (t >> 4) << 4;
        const int j0 = (t & 15) << 4;
        v16h A = {}, Bv = {};
        if (lane < 16) {                      // lanes 0-15 hold K=0..7 (elems 0..7)
            A[0]  = areaH[i0 + lane];         // K=0 : area_i
            A[1]  = (_Float16)1.0f;           // K=1 : 1
            Bv[0] = (_Float16)1.0f;           // K=0 : 1
            Bv[1] = areaH[j0 + lane];         // K=1 : area_j
        }
        v8f z = {};
        v8f d = __builtin_amdgcn_wmma_f32_16x16x32_f16(
                    false, A, false, Bv, (short)0, z, false, false);
        const int j  = j0 + (lane & 15);
        const int hi = (lane >> 4) << 3;
        const float jy0 = y0s[j], jx0 = x0s[j], jy1 = y1s[j], jx1 = x1s[j];
#pragma unroll
        for (int r = 0; r < 8; ++r) {
            const int i = i0 + hi + r;
            float iy = fmaxf(0.0f, fminf(y1s[i], jy1) - fmaxf(y0s[i], jy0));
            float ix = fmaxf(0.0f, fminf(x1s[i], jx1) - fmaxf(x0s[i], jx0));
            float inter = iy * ix;
            float uni = fmaxf(d[r] - inter, 1e-9f);     // area_i+area_j-inter
            if (inter > NMS_T * uni)
                atomicOr(&adj[i][j >> 5], 1u << (j & 31));
        }
    }
    __syncthreads();

    // serial greedy suppression on bitmasks (j > i only)
    if (tid == 0) {
        for (int w = 0; w < 8; ++w) {
            unsigned m = 0u;
            for (int x = 0; x < 32; ++x)
                if (scs[w * 32 + x] > 0.0f) m |= (1u << x);
            keepm[w] = m;
        }
        for (int i = 0; i < KSEL; ++i) {
            if ((keepm[i >> 5] >> (i & 31)) & 1u) {
                const int wi = i >> 5;
                for (int w = wi; w < 8; ++w) {
                    unsigned m = adj[i][w];
                    if (w == wi) {
                        const unsigned low = ((i & 31) == 31)
                            ? 0xffffffffu : ((1u << ((i & 31) + 1)) - 1u);
                        m &= ~low;
                    }
                    keepm[w] &= ~m;
                }
            }
        }
    }
    __syncthreads();
    const bool kept = (keepm[tid >> 5] >> (tid & 31)) & 1u;
    cand_val[base + tid] = kept ? scs[tid] : 0.0f;      // in-place
}

// ------------------------------- K4: final top-100 -------------------------
__global__ __launch_bounds__(256) void final_kernel(
        const float* __restrict__ cand_val, const float* __restrict__ cand_box,
        float* __restrict__ out) {
    __shared__ unsigned hist[1024];
    __shared__ float    selv[128];
    __shared__ int      seli[128];
    __shared__ unsigned cnt_hi, nvalid;
    __shared__ int      s_T, s_C1;

    const int b = blockIdx.x, tid = threadIdx.x;
    const int M = NCLS * KSEL;                         // 20480
    const float* vals = cand_val + (size_t)b * M;

    for (int i = tid; i < 1024; i += 256) hist[i] = 0u;
    if (tid == 0) { cnt_hi = 0u; nvalid = 0u; }
    __syncthreads();

    for (int e = tid; e < M; e += 256) {
        int bin = (int)(vals[e] * 1024.0f);
        bin = bin < 0 ? 0 : (bin > 1023 ? 1023 : bin);
        atomicAdd(&hist[bin], 1u);
    }
    __syncthreads();
    if (tid == 0) {
        unsigned acc = 0; int T = 0, C1 = 0;
        for (int t = 1023; t >= 0; --t) {
            unsigned h = hist[t];
            if (acc + h >= KEEP) { T = t; C1 = (int)acc; break; }
            acc += h;
        }
        s_T = T; s_C1 = C1;
    }
    __syncthreads();
    const int T = s_T, C1 = s_C1, need = KEEP - C1;
    for (int i = tid; i < 128; i += 256) { selv[i] = -1.0f; seli[i] = 0; }
    __syncthreads();

    for (int e = tid; e < M; e += 256) {
        float v = vals[e];
        int bin = (int)(v * 1024.0f);
        bin = bin < 0 ? 0 : (bin > 1023 ? 1023 : bin);
        if (bin > T) { unsigned s = atomicAdd(&cnt_hi, 1u); selv[s] = v; seli[s] = e; }
    }
    __syncthreads();
    if (tid == 0) {
        int t = 0;
        for (int e = 0; e < M && t < need; ++e) {
            float v = vals[e];
            int bin = (int)(v * 1024.0f);
            bin = bin < 0 ? 0 : (bin > 1023 ? 1023 : bin);
            if (bin == T) { selv[C1 + t] = v; seli[C1 + t] = e; ++t; }
        }
    }
    __syncthreads();

    for (int k = 2; k <= 128; k <<= 1)
        for (int j = k >> 1; j > 0; j >>= 1) {
            __syncthreads();
            if (tid < 128) {
                int ixj = tid ^ j;
                if (ixj > tid) {
                    bool desc = ((tid & k) == 0);
                    float va = selv[tid], vb = selv[ixj];
                    int   ia = seli[tid], ib = seli[ixj];
                    bool swp = desc ? ((va < vb) || (va == vb && ia > ib))
                                    : ((va > vb) || (va == vb && ia < ib));
                    if (swp) { selv[tid] = vb; selv[ixj] = va;
                               seli[tid] = ib; seli[ixj] = ia; }
                }
            }
        }
    __syncthreads();

    float* nbp = out;                          // [8][100][4]
    float* ncp = out + NB * KEEP * 4;          // [8][100]
    float* nsp = ncp + NB * KEEP;              // [8][100]
    float* ndp = nsp + NB * KEEP;              // [8]
    if (tid < KEEP) {
        const float  s = selv[tid];
        const int    f = seli[tid];
        const bool   valid = s > 0.0f;
        const size_t ob = (size_t)b * KEEP + tid;
        if (valid) {
            const size_t cb = ((size_t)b * M + f) * 4;
            nbp[ob * 4 + 0] = fminf(fmaxf(cand_box[cb + 0], 0.0f), 1.0f);
            nbp[ob * 4 + 1] = fminf(fmaxf(cand_box[cb + 1], 0.0f), 1.0f);
            nbp[ob * 4 + 2] = fminf(fmaxf(cand_box[cb + 2], 0.0f), 1.0f);
            nbp[ob * 4 + 3] = fminf(fmaxf(cand_box[cb + 3], 0.0f), 1.0f);
            ncp[ob] = (float)(f >> 8);         // class = flat / 256
            nsp[ob] = s;
            atomicAdd(&nvalid, 1u);
        } else {
            nbp[ob * 4 + 0] = 0.0f; nbp[ob * 4 + 1] = 0.0f;
            nbp[ob * 4 + 2] = 0.0f; nbp[ob * 4 + 3] = 0.0f;
            ncp[ob] = 0.0f; nsp[ob] = 0.0f;
        }
    }
    __syncthreads();
    if (tid == 0) ndp[b] = (float)nvalid;      // sum(ceil(ns)) == #(s>0)
}

// ------------------------------- launch ------------------------------------
extern "C" void kernel_launch(void* const* d_in, const int* in_sizes, int n_in,
                              void* d_out, int out_size, void* d_ws, size_t ws_size,
                              hipStream_t stream) {
    (void)in_sizes; (void)n_in; (void)out_size; (void)ws_size;
    const float* x3 = (const float*)d_in[0];
    const float* x4 = (const float*)d_in[1];
    const float* x5 = (const float*)d_in[2];
    float* out = (float*)d_out;

    // workspace carve (floats); total 17,753,600 f32 ~= 71 MB
    float* ws       = (float*)d_ws;
    float* boxes    = ws;                                       // B*N*4
    float* clsT     = boxes + (size_t)NB * NPTS * 4;            // B*C*N (transposed)
    float* cand_val = clsT + (size_t)NB * NCLS * NPTS;          // B*C*K
    float* cand_box = cand_val + (size_t)NB * NCLS * KSEL;      // B*C*K*4

    const int total = NB * NPTS;
    decode_kernel<<<(total + 255) / 256, 256, 0, stream>>>(x3, x4, x5, boxes, clsT);
    topk_kernel  <<<NB * NCLS, 256, 0, stream>>>(clsT, boxes, cand_val, cand_box);
    nms_kernel   <<<NB * NCLS, 256, 0, stream>>>(cand_val, cand_box);
    final_kernel <<<NB, 256, 0, stream>>>(cand_val, cand_box, out);
}
